// QLSTM_65481071410555
// MI455X (gfx1250) — matrix operand order repaired and checked by reading, not compiled, not run on hardware
//
#include <hip/hip_runtime.h>
#include <hip/hip_bf16.h>
#include <math.h>

// ---------------------------------------------------------------------------
// QLSTM for MI455X (gfx1250, wave32).
//
// Phase 1 (parallel, WMMA): preX[T*B,16] = X[T*B,128] * W[:, :128]^T
//   v_wmma_f32_16x16x32_f16, f32 accumulate. Bandwidth-bound (128 MB read).
// Phase 2 (sequential over T): closed-form quantum circuit
//   z0=C1C2C3, z1=C0C1, z2=C0C1C2, z3=C0C1C2C3 with Cw=cos(pre_w),
//   plus LSTM cell. One thread per batch element.
// ---------------------------------------------------------------------------

typedef __attribute__((ext_vector_type(16))) _Float16 v16h;
typedef __attribute__((ext_vector_type(8)))  float    v8f;

#define QT 128   // timesteps
#define QB 2048  // batch
#define QD 128   // input dim
#define QDH 132  // D + H
#define TB (QT * QB)

// ---------------------------------------------------------------------------
// Kernel 1: preX[r][n] = sum_k X[r][k] * W[n][k],  r in [0, T*B), n in [0,16)
// One wave handles one 16-row M tile, K = 128 via 4 x wmma_f32_16x16x32_f16.
// ---------------------------------------------------------------------------
__global__ __launch_bounds__(256) void qlstm_gemm_wmma(
    const float* __restrict__ X,    // [T*B, 128]
    const float* __restrict__ W,    // [16, 132] row-major (first 128 cols used)
    float* __restrict__ preX)       // [T*B, 16]
{
  const int lane = threadIdx.x & 31;
  const int wave = threadIdx.x >> 5;
  const int tile = blockIdx.x * 8 + wave;   // 16384 tiles total
  const int half = lane >> 4;               // lane half (0: lanes 0-15, 1: 16-31)
  const int l16  = lane & 15;

  const int row = tile * 16 + l16;                    // A-matrix row M = l16
  const float* __restrict__ xrow = X + (size_t)row * QD;
  const float* __restrict__ wrow = W + (size_t)l16 * QDH;  // B column N = l16

  v8f acc = {};
  #pragma unroll
  for (int kb = 0; kb < QD; kb += 32) {
    // A (16-bit 16x32 layout): lanes 0-15 hold K = kb+{0..7, 16..23},
    //                          lanes 16-31 hold K = kb+{8..15, 24..31}
    v16h a, b;
    const float* pa0 = xrow + kb + half * 8;        // 8 contiguous floats
    const float* pa1 = pa0 + 16;                    // 8 contiguous floats
    #pragma unroll
    for (int p = 0; p < 8; ++p) {
      a[p]     = (_Float16)pa0[p];
      a[p + 8] = (_Float16)pa1[p];
    }
    // B (32x16): lanes 0-15 hold K = kb+{0..7, 16..23}, lanes 16-31 hold K = kb+{8..15, 24..31},
    // column N = l16.  B[K][N] = W[N][K]  (from W row l16).
    const float* pb0 = wrow + kb + half * 8;
    const float* pb1 = pb0 + 16;
    #pragma unroll
    for (int p = 0; p < 8; ++p) {
      b[p]     = (_Float16)pb0[p];
      b[p + 8] = (_Float16)pb1[p];
    }

    acc = __builtin_amdgcn_wmma_f32_16x16x32_f16(
        /*neg_a=*/false, a, /*neg_b=*/false, b,
        /*c_mod=*/(short)0, acc, /*reuse_a=*/false, /*reuse_b=*/false);
  }

  // C/D layout: VGPR v -> row M = v + half*8, column N = l16.
  float* outbase = preX + (size_t)tile * 16 * 16;
  #pragma unroll
  for (int v = 0; v < 8; ++v)
    outbase[(size_t)(v + half * 8) * 16 + l16] = acc[v];
}

// ---------------------------------------------------------------------------
// Kernel 2: sequential recurrence. One thread per batch element b.
// ---------------------------------------------------------------------------
__device__ __forceinline__ float fast_sigmoid(float x) {
  return 1.0f / (1.0f + __expf(-x));
}
__device__ __forceinline__ float fast_tanh(float x) {
  // safe for any magnitude: exp of a non-positive argument only
  float e = __expf(-2.0f * fabsf(x));
  float t = (1.0f - e) / (1.0f + e);
  return copysignf(t, x);
}

__global__ __launch_bounds__(256) void qlstm_recur(
    const float* __restrict__ preX,  // [T*B, 16]
    const float* __restrict__ W,     // [16, 132]; cols 128..131 = recurrent part
    const float* __restrict__ bg,    // [16] b_gates
    const float* __restrict__ th,    // [16] theta
    float* __restrict__ out)         // outputs [T,B,4] ++ hx [B,4] ++ cx [B,4]
{
  const int b = blockIdx.x * blockDim.x + threadIdx.x;
  if (b >= QB) return;

  float wh[16][4];
  float bias[16];
  #pragma unroll
  for (int gq = 0; gq < 16; ++gq) {
    const float* wr = W + (size_t)gq * QDH + QD;
    wh[gq][0] = wr[0]; wh[gq][1] = wr[1]; wh[gq][2] = wr[2]; wh[gq][3] = wr[3];
    bias[gq] = bg[gq] + th[gq];
  }

  float h[4] = {0.f, 0.f, 0.f, 0.f};
  float c[4] = {0.f, 0.f, 0.f, 0.f};

  for (int t = 0; t < QT; ++t) {
    const float* pr = preX + ((size_t)t * QB + b) * 16;

    float co[16];
    #pragma unroll
    for (int gq = 0; gq < 16; ++gq) {
      float pre = pr[gq] + bias[gq]
                + wh[gq][0] * h[0] + wh[gq][1] * h[1]
                + wh[gq][2] * h[2] + wh[gq][3] * h[3];
      co[gq] = __cosf(pre);
    }

    // closed-form circuit: z0=C1C2C3, z1=C0C1, z2=C0C1C2, z3=C0C1C2C3
    float z[4][4];
    #pragma unroll
    for (int g = 0; g < 4; ++g) {
      float C0 = co[g * 4 + 0], C1 = co[g * 4 + 1];
      float C2 = co[g * 4 + 2], C3 = co[g * 4 + 3];
      float t12 = C1 * C2;
      z[g][0] = t12 * C3;
      z[g][1] = C0 * C1;
      z[g][2] = C0 * t12;
      z[g][3] = z[g][2] * C3;
    }

    #pragma unroll
    for (int j = 0; j < 4; ++j) {
      float fj = fast_sigmoid(z[0][j]);
      float ij = fast_sigmoid(z[1][j]);
      float gj = fast_tanh(z[2][j]);
      float oj = fast_sigmoid(z[3][j]);
      c[j] = fj * c[j] + ij * gj;
      h[j] = oj * fast_tanh(c[j]);
    }

    float4* dst = (float4*)(out + ((size_t)t * QB + b) * 4);
    *dst = make_float4(h[0], h[1], h[2], h[3]);
  }

  float4* hxo = (float4*)(out + (size_t)QT * QB * 4 + (size_t)b * 4);
  float4* cxo = (float4*)(out + (size_t)QT * QB * 4 + (size_t)QB * 4 + (size_t)b * 4);
  *hxo = make_float4(h[0], h[1], h[2], h[3]);
  *cxo = make_float4(c[0], c[1], c[2], c[3]);
}

// ---------------------------------------------------------------------------
extern "C" void kernel_launch(void* const* d_in, const int* in_sizes, int n_in,
                              void* d_out, int out_size, void* d_ws, size_t ws_size,
                              hipStream_t stream) {
  const float* X  = (const float*)d_in[0];  // [128, 2048, 128]
  const float* W  = (const float*)d_in[1];  // [4, 4, 132]
  const float* bg = (const float*)d_in[2];  // [4, 4]
  const float* th = (const float*)d_in[3];  // [4, 4]
  float* preX = (float*)d_ws;               // [T*B, 16] = 16.8 MB

  // Phase 1: 16384 M-tiles of 16 rows, 8 waves (tiles) per 256-thread block.
  qlstm_gemm_wmma<<<2048, 256, 0, stream>>>(X, W, preX);
  // Phase 2: one thread per batch element.
  qlstm_recur<<<(QB + 255) / 256, 256, 0, stream>>>(preX, W, bg, th, (float*)d_out);
}
